// MoleculePoolingLayer_34926674051616
// MI455X (gfx1250) — compile-verified, matrix operand order checked by reading, not run on hardware
//
#include <hip/hip_runtime.h>

#define IN_DIM 256
#define HID 64
#define OUT_DIM 16
#define NMOL 65536
#define SEGS 8192

typedef __attribute__((ext_vector_type(16))) __bf16 v16bf;
typedef __attribute__((ext_vector_type(8)))  float  v8f;

struct __attribute__((aligned(16))) U4 { unsigned int x, y, z, w; };
union Frag { v16bf v; U4 q[2]; };

// hardware f32 -> bf16 convert (v_cvt, RNE), 1 VALU op instead of 4
__device__ __forceinline__ unsigned short f2bf(float f) {
  union { __bf16 b; unsigned short u; } c;
  c.b = (__bf16)f;
  return c.u;
}

__device__ __forceinline__ unsigned int pack2(float a, float b) {
  return (unsigned int)f2bf(a) | ((unsigned int)f2bf(b) << 16);
}

// K index (within a 32-wide K chunk) of fragment element e for this lane,
// per the CDNA5 16-bit A/B VGPR layout (CDNA5 ISA 7.12.2).
__device__ __forceinline__ int kOf(int lane, int e) {
  int p = e >> 1, h = e & 1;
  return ((p & 4) ? 16 : 0) + ((lane & 16) ? 8 : 0) + (p & 3) * 2 + h;
}

// B fragment pre-packed in ws: lane's 16 bf16 contiguous -> two global b128 loads
__device__ __forceinline__ v16bf load_frag_g(const unsigned short* p, int lane) {
  Frag f;
  const U4* q = (const U4*)(p + lane * 16);
  f.q[0] = q[0];
  f.q[1] = q[1];
  return f.v;
}

// A fragment from row-major bf16 LDS tile: two ds_load_b128 per lane
__device__ __forceinline__ v16bf load_a_lds(const unsigned short* row, int lane, int ks) {
  int b0 = ks * 32 + ((lane & 16) ? 8 : 0);
  Frag f;
  f.q[0] = *(const U4*)(row + b0);
  f.q[1] = *(const U4*)(row + b0 + 16);
  return f.v;
}

// ---------------- kernel 1: repack weights to WMMA B-fragment order ----------------
// ws layout (bf16 elements): wfB[196608] | w1B[49152] | w2B[3072]
__global__ void pack_weights_kernel(
    const float* __restrict__ Wf,
    const float* __restrict__ W1r, const float* __restrict__ W1a, const float* __restrict__ W1p,
    const float* __restrict__ W2r, const float* __restrict__ W2a, const float* __restrict__ W2p,
    unsigned short* __restrict__ ws16) {
  int t = blockIdx.x * 256 + threadIdx.x;
  unsigned short* wfB = ws16;
  unsigned short* w1B = ws16 + 196608;
  unsigned short* w2B = ws16 + 245760;
  if (t < 196608) {            // Wf: [role][jtile16][kstep8][lane32][e16]
    int e = t & 15, lane = (t >> 4) & 31, ks = (t >> 9) & 7, jt = (t >> 12) & 15, role = t >> 16;
    int k = ks * 32 + kOf(lane, e);
    int n = jt * 16 + (lane & 15);
    wfB[t] = f2bf(Wf[(size_t)(role * 256 + k) * 256 + n]);
  } else if (t < 245760) {     // W1: [role][ntile4][kstep8][lane32][e16]
    int i = t - 196608;
    int e = i & 15, lane = (i >> 4) & 31, ks = (i >> 9) & 7, nt = (i >> 12) & 3, role = i >> 14;
    const float* W1 = role == 0 ? W1r : (role == 1 ? W1a : W1p);
    int k = ks * 32 + kOf(lane, e);
    int n = nt * 16 + (lane & 15);
    w1B[i] = f2bf(W1[k * HID + n]);
  } else if (t < 248832) {     // W2: [role][kstep2][lane32][e16]
    int i = t - 245760;
    int e = i & 15, lane = (i >> 4) & 31, ks = (i >> 9) & 1, role = i >> 10;
    const float* W2 = role == 0 ? W2r : (role == 1 ? W2a : W2p);
    int k = ks * 32 + kOf(lane, e);
    w2B[i] = f2bf(W2[k * OUT_DIM + (lane & 15)]);
  }
}

// ---------------- kernel 2: fully fused ----------------
// Per block: 2 segments (16 molecules per role). For each role:
//   stage X -> LDS bf16; waves 0-3 compute one hidden n-tile each (X@W1+b1, relu);
//   all waves compute Y = X@Wf_role column tiles (rebalanced 1/1/1/1/3/3/3/3) and
//   deposit Y^T into LDS in stage-2 B-fragment layout; wave 0 runs H@W2+b2 -> sigmoid
//   and parks w^T in LDS in stage-2 A-fragment layout.
// Then one WMMA per (segment, jtile): out[s] = w^T[16,24] @ Y[24,256] + bf.
__global__ void __launch_bounds__(256) fused_kernel(
    const float* __restrict__ rx, const float* __restrict__ ax, const float* __restrict__ px,
    const float* __restrict__ b1r, const float* __restrict__ b1a, const float* __restrict__ b1p,
    const float* __restrict__ b2r, const float* __restrict__ b2a, const float* __restrict__ b2p,
    const float* __restrict__ bfv, const unsigned short* __restrict__ ws16,
    float* __restrict__ out) {
  __shared__ unsigned short Xs[16 * 264];        // 16 molecule rows, bf16, stride 264
  __shared__ unsigned short Ys[2 * 256 * 48];    // [seg][col 0..255][kslot 0..31 (+pad)]
  __shared__ unsigned short Hs[16 * 72];         // hidden 16x64 bf16, stride 72
  __shared__ unsigned short Ws[3 * 16 * 16];     // gates w^T: [role][out][mol16]
  const unsigned short* wfB = ws16;
  const unsigned short* w1B = ws16 + 196608;
  const unsigned short* w2B = ws16 + 245760;
  int wave = threadIdx.x >> 5, lane = threadIdx.x & 31;
  int s0 = blockIdx.x * 2;                       // block handles segments s0, s0+1

  // zero kslots 24..31 of every Y row (stage-2 A is zero there; avoid 0*NaN)
  for (int i = threadIdx.x; i < 512; i += 256) {
    U4 z = {0, 0, 0, 0};
    *(U4*)&Ys[i * 48 + 24] = z;
  }

  int m    = lane & 15;
  int m0   = (lane & 16) ? 8 : 0;
  int seg1 = (lane & 16) ? 1 : 0;

  for (int role = 0; role < 3; ++role) {
    const float* X  = role == 0 ? rx  : (role == 1 ? ax  : px);
    const float* b1 = role == 0 ? b1r : (role == 1 ? b1a : b1p);
    const float* b2 = role == 0 ? b2r : (role == 1 ? b2a : b2p);

    __syncthreads();                             // prior-iter Xs/Hs reads done
    size_t rowbase = (size_t)s0 * 8;             // 16 molecules = 2 segments of 8
    for (int i = 0; i < 4; ++i) {
      int flat = i * 256 + threadIdx.x;          // 1024 float4 total
      int row = flat >> 6, c4 = flat & 63;
      const float4 v = *(const float4*)(X + (rowbase + row) * IN_DIM + c4 * 4);
      unsigned int* d = (unsigned int*)&Xs[row * 264 + c4 * 4];
      d[0] = pack2(v.x, v.y);
      d[1] = pack2(v.z, v.w);
    }
    __syncthreads();                             // X staged

    // gate hidden: waves 0-3, one 16-col n-tile each: relu(X@W1 + b1)
    if (wave < 4) {
      int nt = wave;
      v8f acc = {};
      for (int ks = 0; ks < 8; ++ks) {
        v16bf a = load_a_lds(&Xs[m * 264], lane, ks);
        v16bf b = load_frag_g(w1B + (size_t)((role * 4 + nt) * 8 + ks) * 512, lane);
        acc = __builtin_amdgcn_wmma_f32_16x16x32_bf16(false, a, false, b, (short)0, acc, false, false);
      }
      float bias = b1[nt * 16 + m];
      for (int v = 0; v < 8; ++v)
        Hs[(v + m0) * 72 + nt * 16 + m] = f2bf(fmaxf(acc[v] + bias, 0.f));
    }

    // Y = X @ Wf_role, column tiles rebalanced across waves
    int jbeg = (wave < 4) ? wave : 4 + (wave - 4) * 3;
    int jcnt = (wave < 4) ? 1 : 3;
    for (int jj = 0; jj < jcnt; ++jj) {
      int j = jbeg + jj;
      v8f acc = {};
      for (int ks = 0; ks < 8; ++ks) {
        v16bf a = load_a_lds(&Xs[m * 264], lane, ks);
        v16bf b = load_frag_g(wfB + (size_t)((role * 16 + j) * 8 + ks) * 512, lane);
        acc = __builtin_amdgcn_wmma_f32_16x16x32_bf16(false, a, false, b, (short)0, acc, false, false);
      }
      // lane holds col n of 8 consecutive molecules of one segment -> one ds_store_b128
      int n = j * 16 + m;
      U4 pk;
      pk.x = pack2(acc[0], acc[1]);
      pk.y = pack2(acc[2], acc[3]);
      pk.z = pack2(acc[4], acc[5]);
      pk.w = pack2(acc[6], acc[7]);
      *(U4*)&Ys[(seg1 * 256 + n) * 48 + role * 8] = pk;   // kslots role*8 .. role*8+7
    }
    __syncthreads();                             // Hs ready

    // gate output: wave 0 only: w = sigmoid(H@W2 + b2), park w^T in LDS
    if (wave == 0) {
      v8f acc2 = {};
      for (int ks = 0; ks < 2; ++ks) {
        v16bf a = load_a_lds(&Hs[m * 72], lane, ks);
        v16bf b = load_frag_g(w2B + (size_t)(role * 2 + ks) * 512, lane);
        acc2 = __builtin_amdgcn_wmma_f32_16x16x32_bf16(false, a, false, b, (short)0, acc2, false, false);
      }
      float bias2 = b2[m];
      for (int v = 0; v < 8; ++v) {
        float s = 1.f / (1.f + __expf(-(acc2[v] + bias2)));
        Ws[(role * 16 + m) * 16 + (v + m0)] = f2bf(s);    // [role][out=m][mol=v+m0]
      }
    }
  }
  __syncthreads();                               // Ys + Ws complete

  // phase 2: out[s][:, jtile] = A_s[16,32] @ Ys[s][:,jtile]  (K slots: r=0-7, a=8-15, p=16-23)
  int seg   = wave & 1;
  int jbase = (wave >> 1) * 4;
  Frag af;
  {
    int r0 = (lane & 16) ? 1 : 0;                // lanes<16: k 0-7 & 16-23; lanes>=16: k 8-15 & 24-31
    af.q[0] = *(const U4*)&Ws[(r0 * 16 + m) * 16 + seg * 8];
    if (lane & 16) {
      U4 z = {0, 0, 0, 0};
      af.q[1] = z;                               // k 24-31 zero padding
    } else {
      af.q[1] = *(const U4*)&Ws[(2 * 16 + m) * 16 + seg * 8];
    }
  }
  for (int jj = 0; jj < 4; ++jj) {
    int j = jbase + jj;
    int n = j * 16 + m;
    Frag bfrag;
    const unsigned short* yr = &Ys[(seg * 256 + n) * 48];
    int kb = (lane & 16) ? 8 : 0;
    bfrag.q[0] = *(const U4*)(yr + kb);
    bfrag.q[1] = *(const U4*)(yr + 16 + kb);
    v8f d = {};
    d = __builtin_amdgcn_wmma_f32_16x16x32_bf16(false, af.v, false, bfrag.v, (short)0, d, false, false);
    float bias = bfv[n];
    float* o = out + ((size_t)(s0 + seg) * 16 + m0) * IN_DIM + n;
    for (int v = 0; v < 8; ++v)
      o[(size_t)v * IN_DIM] = d[v] + bias;
  }
}

extern "C" void kernel_launch(void* const* d_in, const int* in_sizes, int n_in,
                              void* d_out, int out_size, void* d_ws, size_t ws_size,
                              hipStream_t stream) {
  const float* rx  = (const float*)d_in[0];
  const float* ax  = (const float*)d_in[2];
  const float* px  = (const float*)d_in[4];
  const float* W1r = (const float*)d_in[7];
  const float* b1r = (const float*)d_in[8];
  const float* W2r = (const float*)d_in[9];
  const float* b2r = (const float*)d_in[10];
  const float* W1a = (const float*)d_in[11];
  const float* b1a = (const float*)d_in[12];
  const float* W2a = (const float*)d_in[13];
  const float* b2a = (const float*)d_in[14];
  const float* W1p = (const float*)d_in[15];
  const float* b1p = (const float*)d_in[16];
  const float* W2p = (const float*)d_in[17];
  const float* b2p = (const float*)d_in[18];
  const float* Wf  = (const float*)d_in[19];
  const float* bfv = (const float*)d_in[20];

  unsigned short* ws16 = (unsigned short*)d_ws;
  float* out = (float*)d_out;

  pack_weights_kernel<<<972, 256, 0, stream>>>(Wf, W1r, W1a, W1p, W2r, W2a, W2p, ws16);
  fused_kernel<<<SEGS / 2, 256, 0, stream>>>(rx, ax, px, b1r, b1a, b1p,
                                             b2r, b2a, b2p, bfv, ws16, out);
}
